// LateInteractionModel_79843442032821
// MI455X (gfx1250) — matrix operand order repaired and checked by reading, not compiled, not run on hardware
//
#include <hip/hip_runtime.h>
#include <hip/hip_bf16.h>

// Problem constants (match reference)
#define BB  64
#define STT 512
#define SLL 32
#define LL  5
#define HH  768
#define DD  128
#define NEGV (-1e9f)

typedef __attribute__((ext_vector_type(16))) __bf16 v16bf;
typedef __attribute__((ext_vector_type(8)))  __bf16 v8bf;
typedef __attribute__((ext_vector_type(8)))  float  v8f;
typedef __attribute__((ext_vector_type(4)))  float  v4f;

// f32 -> bf16: native fptrunc (RNE). gfx1250 has native bf16; the compiler
// should pair these into v_cvt_pk_bf16_f32 instead of the 3-op bit sequence.
__device__ __forceinline__ __bf16 f2bf(float f) {
    return (__bf16)f;
}

__device__ __forceinline__ v16bf cat16(v8bf lo, v8bf hi) {
    return __builtin_shufflevector(lo, hi, 0,1,2,3,4,5,6,7,8,9,10,11,12,13,14,15);
}

// ---------------------------------------------------------------------------
// Kernel 0: one-time W transpose + bf16 convert: Wt[d][k] = bf16(W[k][d]).
// 196 KB result stays L2-resident (192 MB L2) for all projection blocks.
// ---------------------------------------------------------------------------
__global__ __launch_bounds__(256) void wt_kernel(const float* __restrict__ W,
                                                 __bf16* __restrict__ Wt) {
    int i = blockIdx.x * 256 + threadIdx.x;   // grid covers H*D exactly
    int k = i >> 7, d = i & 127;              // read coalesced over d
    Wt[(long)d * HH + k] = f2bf(W[i]);
}

// ---------------------------------------------------------------------------
// Kernel 1: Y[r,:] = normalize(X[r,:] @ W + b), bf16 out. No LDS, no barriers.
// Block = 256 threads (8 waves); each wave owns 16 rows x all 128 cols:
// 8 accumulators, 8 WMMAs per K-chunk sharing one A fragment. A comes from
// global f32 (converted once in-register); B fragments are contiguous 16B
// loads from the L2-resident transposed bf16 Wt.
// ---------------------------------------------------------------------------
__global__ __launch_bounds__(256) void proj_norm_kernel(
    const float* __restrict__ X, const __bf16* __restrict__ Wt,
    const float* __restrict__ bias, __bf16* __restrict__ Y) {

    const int tid  = threadIdx.x;
    const int lane = tid & 31;
    const int wave = tid >> 5;
    const int l15  = lane & 15;
    const int hi   = lane >> 4;

    const long rowbase = (long)blockIdx.x * 128 + wave * 16;
    const float* xr = X + (rowbase + l15) * HH;   // A-matrix row for this lane

    float bv[8];
#pragma unroll
    for (int t = 0; t < 8; ++t) bv[t] = bias[t * 16 + l15];

    v8f c[8] = {};
    for (int kc = 0; kc < HH / 32; ++kc) {
        const int ka = kc * 32 + hi * 8;
        if (kc + 1 < HH / 32)
            __builtin_prefetch(xr + (kc + 1) * 32 + hi * 8, 0, 1); // global_prefetch_b8

        // A fragment: e0..7 -> k=ka..ka+7 ; e8..15 -> k=ka+16..ka+23
        v4f x0 = *(const v4f*)(xr + ka);
        v4f x1 = *(const v4f*)(xr + ka + 4);
        v4f x2 = *(const v4f*)(xr + ka + 16);
        v4f x3 = *(const v4f*)(xr + ka + 20);
        v16bf a;
#pragma unroll
        for (int e = 0; e < 4; ++e) {
            a[e]      = f2bf(x0[e]);
            a[4 + e]  = f2bf(x1[e]);
            a[8 + e]  = f2bf(x2[e]);
            a[12 + e] = f2bf(x3[e]);
        }

        // B fragments: lane col d = t*16+l15; e0..7 -> k=kc*32+hi*8+e ; e8..15 -> k=kc*32+hi*8+16+(e-8)
#pragma unroll
        for (int t = 0; t < 8; ++t) {
            const __bf16* wp = Wt + (long)(t * 16 + l15) * HH + kc * 32 + hi * 8;
            v16bf bf = cat16(*(const v8bf*)wp, *(const v8bf*)(wp + 16));
            c[t] = __builtin_amdgcn_wmma_f32_16x16x32_bf16(
                       false, a, false, bf, (short)0, c[t], false, false);
        }
    }

    // bias add (C layout: M = r + 8*hi, N(col within tile) = l15)
#pragma unroll
    for (int t = 0; t < 8; ++t)
#pragma unroll
        for (int r = 0; r < 8; ++r) c[t][r] += bv[t];

    // row L2 norms: per-lane partial over its 8 cols, then xor-reduce the
    // 16 lanes of each half (halves hold disjoint row sets)
    float inv[8];
#pragma unroll
    for (int r = 0; r < 8; ++r) {
        float ss = 0.f;
#pragma unroll
        for (int t = 0; t < 8; ++t) { float v = c[t][r]; ss += v * v; }
        for (int off = 1; off <= 8; off <<= 1)
            ss += __shfl_xor(ss, off, 32);
        inv[r] = 1.0f / fmaxf(sqrtf(ss), 1e-12f);
    }

    // store normalized bf16 (lanes of a half write 32 contiguous bytes)
#pragma unroll
    for (int r = 0; r < 8; ++r) {
        const long row = rowbase + r + 8 * hi;
#pragma unroll
        for (int t = 0; t < 8; ++t)
            Y[row * DD + t * 16 + l15] = f2bf(c[t][r] * inv[r]);
    }
}

// ---------------------------------------------------------------------------
// Kernel 2: per (b,l): sim = T[b] @ Lab[b,l]^T [512x32], masked max over SL,
// text-masked mean over ST, sigmoid. Block = 256 threads (8 waves), one block
// per (b,l). Label tile (32x128 bf16 = 8KB) stays in LDS as the B operand.
// ---------------------------------------------------------------------------
__global__ __launch_bounds__(256) void maxsim_kernel(
    const __bf16* __restrict__ T, const __bf16* __restrict__ Lab,
    const float* __restrict__ text_mask, const int* __restrict__ label_mask,
    float* __restrict__ out) {

    __shared__ __bf16 sLab[SLL][DD];   // (s, d); B[k][n] = sLab[n][k]
    __shared__ int    sLM[SLL];
    __shared__ float  sRed[256];
    __shared__ float  sDen[256];

    const int tid  = threadIdx.x;
    const int lane = tid & 31;
    const int wave = tid >> 5;
    const int l15  = lane & 15;
    const int hi   = lane >> 4;
    const int b    = blockIdx.x / LL;
    const int l    = blockIdx.x % LL;

    const long labBase = (long)(b * LL + l) * SLL * DD;
    for (int i = tid; i < SLL * DD; i += 256)
        (&sLab[0][0])[i] = Lab[labBase + i];
    if (tid < SLL) sLM[tid] = label_mask[(b * LL + l) * SLL + tid];
    __syncthreads();

    float sum_local = 0.f;

    for (int rt = wave; rt < STT / 16; rt += 8) {
        const __bf16* ta = T + ((long)b * STT + rt * 16 + l15) * DD;
        if (rt + 8 < STT / 16)
            __builtin_prefetch(T + ((long)b * STT + (rt + 8) * 16 + l15) * DD, 0, 1);
        v8f c0 = {}, c1 = {};
#pragma unroll
        for (int kc = 0; kc < DD / 32; ++kc) {
            const int ka = kc * 32 + hi * 8;
            v16bf a = cat16(*(const v8bf*)(ta + ka),
                            *(const v8bf*)(ta + ka + 16));
            const int kb = kc * 32 + hi * 8;
            v16bf bf0 = cat16(*(const v8bf*)&sLab[l15][kb],
                              *(const v8bf*)&sLab[l15][kb + 16]);
            v16bf bf1 = cat16(*(const v8bf*)&sLab[16 + l15][kb],
                              *(const v8bf*)&sLab[16 + l15][kb + 16]);
            c0 = __builtin_amdgcn_wmma_f32_16x16x32_bf16(
                     false, a, false, bf0, (short)0, c0, false, false);
            c1 = __builtin_amdgcn_wmma_f32_16x16x32_bf16(
                     false, a, false, bf1, (short)0, c1, false, false);
        }
        // masked max over SL: c0 col s=l15, c1 col s=16+l15; row = rt*16+r+8*hi
        const bool m0 = sLM[l15] != 0;
        const bool m1 = sLM[16 + l15] != 0;
#pragma unroll
        for (int r = 0; r < 8; ++r) {
            float v = fmaxf(m0 ? c0[r] : NEGV, m1 ? c1[r] : NEGV);
            for (int off = 1; off <= 8; off <<= 1)
                v = fmaxf(v, __shfl_xor(v, off, 32));
            if (l15 == 0) {
                int trow = rt * 16 + r + 8 * hi;
                sum_local += v * text_mask[b * STT + trow];
            }
        }
    }

    sRed[tid] = sum_local;
    sDen[tid] = text_mask[b * STT + tid] + text_mask[b * STT + 256 + tid];
    __syncthreads();
    for (int s = 128; s > 0; s >>= 1) {
        if (tid < s) { sRed[tid] += sRed[tid + s]; sDen[tid] += sDen[tid + s]; }
        __syncthreads();
    }
    if (tid == 0) {
        float denom = fmaxf(sDen[0], 1.0f);
        float x = sRed[0] / denom;
        out[b * LL + l] = 1.0f / (1.0f + expf(-x));
        out[BB * LL + b * LL + l] = 1.0f;   // boolean mask output (all true)
    }
}

extern "C" void kernel_launch(void* const* d_in, const int* in_sizes, int n_in,
                              void* d_out, int out_size, void* d_ws, size_t ws_size,
                              hipStream_t stream) {
    (void)in_sizes; (void)n_in; (void)out_size; (void)ws_size;
    const float* text_hidden  = (const float*)d_in[0]; // [B, ST, H]
    const float* label_hidden = (const float*)d_in[1]; // [B, L, SL, H]
    const float* text_mask    = (const float*)d_in[2]; // [B, ST]
    const int*   label_mask   = (const int*)d_in[3];   // [B, L, SL]
    const float* W            = (const float*)d_in[4]; // [H, D]
    const float* bias         = (const float*)d_in[5]; // [D]
    float* out = (float*)d_out;                        // [B*L scores][B*L mask]

    // workspace layout
    __bf16* Tn = (__bf16*)d_ws;                         // [B*ST, D]   8.4 MB
    __bf16* Ln = Tn + (size_t)BB * STT * DD;            // [B*L*SL, D] 2.6 MB
    __bf16* Wt = Ln + (size_t)BB * LL * SLL * DD;       // [D, H]      0.2 MB

    const int textRows  = BB * STT;        // 32768 -> 256 blocks of 128 rows
    const int labelRows = BB * LL * SLL;   // 10240 -> 80 blocks of 128 rows

    wt_kernel<<<(HH * DD) / 256, 256, 0, stream>>>(W, Wt);
    proj_norm_kernel<<<textRows / 128, 256, 0, stream>>>(text_hidden, Wt, bias, Tn);
    proj_norm_kernel<<<labelRows / 128, 256, 0, stream>>>(label_hidden, Wt, bias, Ln);
    maxsim_kernel<<<BB * LL, 256, 0, stream>>>(Tn, Ln, text_mask, label_mask, out);
}
